// VAE_64080912056804
// MI455X (gfx1250) — 8-bit matrix operand order repaired and checked by reading, compiled, not run on hardware
//
#include <hip/hip_runtime.h>
#include <math.h>


#define B_SZ   131072
#define MTILE  64
#define EPS_C  1e-5f

typedef int v8i __attribute__((ext_vector_type(8)));

__global__ __launch_bounds__(256)
void wquant_kernel(const float* __restrict__ w, int N, int K,
                   int Npad, int Kpad,
                   signed char* __restrict__ wq,
                   float* __restrict__ wdeq)
{
    __shared__ float red[256];
    __shared__ float s_scale;
    const int tid = threadIdx.x;
    const int nelem = N * K;
    float s = 0.f;
    for (int i = tid; i < nelem; i += 256) s += fabsf(w[i]);
    red[tid] = s;
    __syncthreads();
    for (int o = 128; o > 0; o >>= 1) {
        if (tid < o) red[tid] += red[tid + o];
        __syncthreads();
    }
    if (tid == 0) {
        float mean = red[0] / (float)nelem;
        float cl   = fmaxf(mean, EPS_C);
        *wdeq   = cl;
        s_scale = 1.0f / cl;
    }
    __syncthreads();
    const float sc = s_scale;
    const int total = Npad * Kpad;
    for (int i = tid; i < total; i += 256) {
        int n = i / Kpad, k = i - n * Kpad;
        float q = 0.f;
        if (n < N && k < K) {
            q = rintf(w[n * K + k] * sc);
            q = fminf(fmaxf(q, -1.f), 1.f);
        }
        wq[i] = (signed char)(int)q;
    }
}

template <int K, int KPAD>
__global__ __launch_bounds__(256)
void bitlinear_kernel(const float* __restrict__ X,
                      const signed char* __restrict__ WQ, int N, int Npad,
                      const float* __restrict__ bias,
                      const float* __restrict__ wdeq_p,
                      float* __restrict__ Y, int leaky)
{
    __shared__ signed char As[MTILE * KPAD];
    __shared__ float       rdq[MTILE];

    constexpr int NITER = (K + 31) / 32;
    constexpr int QITER = KPAD / 32;

    const int tid  = threadIdx.x;
    const int wave = tid >> 5;
    const int lane = tid & 31;
    const size_t row0 = (size_t)blockIdx.x * MTILE;

    for (int rr = 0; rr < 8; ++rr) {
        const int r = wave * 8 + rr;
        const float* xrow = X + (row0 + r) * (size_t)K;
        float xv[NITER];
        float amax = 0.f;
        #pragma unroll
        for (int i = 0; i < NITER; ++i) {
            const int k = lane + i * 32;
            xv[i] = (k < K) ? xrow[k] : 0.f;
            amax = fmaxf(amax, fabsf(xv[i]));
        }
        #pragma unroll
        for (int o = 16; o > 0; o >>= 1) amax = fmaxf(amax, __shfl_xor(amax, o, 32));
        const float cl = fmaxf(amax, EPS_C);
        const float qs = 127.0f / cl;
        if (lane == 0) rdq[r] = cl * (1.0f / 127.0f);
        #pragma unroll
        for (int i = 0; i < QITER; ++i) {
            const int k = lane + i * 32;
            float q = 0.f;
            if (i < NITER && k < K) {
                q = rintf(xv[i < NITER ? i : 0] * qs);
                q = fminf(fmaxf(q, -128.f), 127.f);
            }
            As[r * KPAD + k] = (signed char)(int)q;
        }
    }
    __syncthreads();

    const int ntiles = Npad >> 4;
    const int tiles  = 4 * ntiles;
    const int nloc = lane & 15;
    const int hi   = lane >> 4;
    const float wd = *wdeq_p;

    for (int t = wave; t < tiles; t += 8) {
        const int mt = t & 3;
        const int nt = t >> 2;
        const signed char* Arow = As + (mt * 16 + nloc) * KPAD + hi * 8;
        const signed char* Brow = WQ + (size_t)(nt * 16 + nloc) * KPAD + hi * 8;

        v8i acc = {0, 0, 0, 0, 0, 0, 0, 0};
        #pragma unroll
        for (int k0 = 0; k0 < KPAD; k0 += 64) {
            if (k0 + 64 < KPAD)
                __builtin_prefetch(Brow + k0 + 64, 0, 1);
            v8i a, b;
            a[0] = *(const int*)(Arow + k0 + 0);
            a[1] = *(const int*)(Arow + k0 + 4);
            a[2] = *(const int*)(Arow + k0 + 16);
            a[3] = *(const int*)(Arow + k0 + 20);
            a[4] = *(const int*)(Arow + k0 + 32);
            a[5] = *(const int*)(Arow + k0 + 36);
            a[6] = *(const int*)(Arow + k0 + 48);
            a[7] = *(const int*)(Arow + k0 + 52);
            b[0] = *(const int*)(Brow + k0 + 0);
            b[1] = *(const int*)(Brow + k0 + 4);
            b[2] = *(const int*)(Brow + k0 + 16);
            b[3] = *(const int*)(Brow + k0 + 20);
            b[4] = *(const int*)(Brow + k0 + 32);
            b[5] = *(const int*)(Brow + k0 + 36);
            b[6] = *(const int*)(Brow + k0 + 48);
            b[7] = *(const int*)(Brow + k0 + 52);
            acc = __builtin_amdgcn_wmma_i32_16x16x64_iu8(true, a, true, b, acc,
                                                         false, false);
        }

        const int n = nt * 16 + nloc;
        if (n < N) {
            const float bn = bias[n];
            const int mbase = mt * 16 + hi * 8;
            #pragma unroll
            for (int r = 0; r < 8; ++r) {
                const int m = mbase + r;
                float y = (float)acc[r] * rdq[m] * wd + bn;
                if (leaky) y = (y > 0.f) ? y : 0.2f * y;
                Y[(row0 + m) * (size_t)N + n] = y;
            }
        }
    }
}

__global__ __launch_bounds__(256)
void reparam_kernel(const float* __restrict__ mu, const float* __restrict__ lv,
                    const float* __restrict__ eps, float* __restrict__ z, int n)
{
    int i = blockIdx.x * 256 + threadIdx.x;
    if (i < n) z[i] = mu[i] + eps[i] * expf(0.5f * lv[i]);
}

extern "C" void kernel_launch(void* const* d_in, const int* in_sizes, int n_in,
                              void* d_out, int out_size, void* d_ws, size_t ws_size,
                              hipStream_t stream)
{
    (void)in_sizes; (void)n_in; (void)out_size; (void)ws_size;

    const float* x   = (const float*)d_in[0];
    const float* eps = (const float*)d_in[1];
    const float* lw[9] = {
        (const float*)d_in[2],  (const float*)d_in[3],  (const float*)d_in[4],
        (const float*)d_in[8],  (const float*)d_in[10],
        (const float*)d_in[12], (const float*)d_in[13], (const float*)d_in[14],
        (const float*)d_in[15]
    };
    const float* lb[9] = {
        (const float*)d_in[5],  (const float*)d_in[6],  (const float*)d_in[7],
        (const float*)d_in[9],  (const float*)d_in[11],
        (const float*)d_in[16], (const float*)d_in[17], (const float*)d_in[18],
        (const float*)d_in[19]
    };
    static const int LN[9] = {200, 200, 200,  32,  32, 200, 200, 200, 784};
    static const int LK[9] = {784, 200, 200, 200, 200,  32, 200, 200, 200};

    float* recon = (float*)d_out;
    float* mu    = recon + (size_t)B_SZ * 784;
    float* lv    = mu    + (size_t)B_SZ * 32;

    char* ws = (char*)d_ws;
    size_t off = 0;
    auto carve = [&](size_t bytes) -> char* {
        char* p = ws + off;
        off = (off + bytes + 255) & ~(size_t)255;
        return p;
    };

    signed char* wq[9];
    int Npad[9], Kpad[9];
    for (int i = 0; i < 9; ++i) {
        Npad[i] = (LN[i] + 15) & ~15;
        Kpad[i] = (LK[i] + 63) & ~63;
        wq[i] = (signed char*)carve((size_t)Npad[i] * (size_t)Kpad[i]);
    }
    float* wdeq = (float*)carve(9 * sizeof(float));
    float* act0 = (float*)carve((size_t)B_SZ * 200 * sizeof(float));
    float* act1 = (float*)carve((size_t)B_SZ * 200 * sizeof(float));
    float* zbuf = (float*)carve((size_t)B_SZ * 32  * sizeof(float));

    for (int i = 0; i < 9; ++i)
        wquant_kernel<<<1, 256, 0, stream>>>(lw[i], LN[i], LK[i],
                                             Npad[i], Kpad[i], wq[i], wdeq + i);

    auto bl = [&](int i, const float* X, float* Y, int leaky) {
        const dim3 grid(B_SZ / MTILE);
        switch (LK[i]) {
        case 784:
            bitlinear_kernel<784, 832><<<grid, 256, 0, stream>>>(
                X, wq[i], LN[i], Npad[i], lb[i], wdeq + i, Y, leaky);
            break;
        case 200:
            bitlinear_kernel<200, 256><<<grid, 256, 0, stream>>>(
                X, wq[i], LN[i], Npad[i], lb[i], wdeq + i, Y, leaky);
            break;
        default:
            bitlinear_kernel<32, 64><<<grid, 256, 0, stream>>>(
                X, wq[i], LN[i], Npad[i], lb[i], wdeq + i, Y, leaky);
            break;
        }
    };

    bl(0, x,    act0, 0);
    bl(1, act0, act1, 0);
    bl(2, act1, act0, 0);
    bl(3, act0, mu,   0);
    bl(4, act0, lv,   0);

    reparam_kernel<<<(B_SZ * 32 + 255) / 256, 256, 0, stream>>>(
        mu, lv, eps, zbuf, B_SZ * 32);

    bl(5, zbuf, act0, 1);
    bl(6, act0, act1, 0);
    bl(7, act1, act0, 1);
    bl(8, act0, recon, 0);
}
